// AtmoSphericNeuralOperatorNet_63488206570144
// MI455X (gfx1250) — matrix operand order repaired and checked by reading, not compiled, not run on hardware
//
#include <hip/hip_runtime.h>
#include <hip/hip_bf16.h>

// ---------------- types / helpers ----------------
typedef __attribute__((ext_vector_type(16))) __bf16 v16bf;
typedef __attribute__((ext_vector_type(8)))  float  v8f;
typedef __attribute__((ext_vector_type(4)))  unsigned int v4u;

union BF16Frag { v16bf v; unsigned short u[16]; v4u q[2]; };

__device__ __forceinline__ unsigned short f2bf(float f) {
  unsigned int x = __builtin_bit_cast(unsigned int, f);
  x += 0x7FFFu + ((x >> 16) & 1u);            // round-to-nearest-even
  return (unsigned short)(x >> 16);
}

__device__ __forceinline__ float gelu_tanh(float x) {
  const float k0 = 0.7978845608028654f;       // sqrt(2/pi)
  float t = tanhf(k0 * (x + 0.044715f * x * x * x));
  return 0.5f * x * (1.0f + t);
}

// fragment-order index for a Z/B tile element (row = K index, pl = column 0..15):
// tile stored as [Kpad/32][32 lanes][16 halfwords], so each lane's WMMA operand
// is 32 contiguous bytes -> 2x ds_load_b128.
__device__ __forceinline__ int frag_index(int row, int pl) {
  return (((row >> 5) * 32) + pl + ((row >> 3) & 1) * 16) * 16 + ((row >> 4) & 1) * 8 + (row & 7);
}

// Model constants
#define HFWF 259920   // 361*720
#define HPWP 16200    // 90*180
#define CE 48
#define HD 96
#define COUT 24

// ---------------- fused disco-conv + WMMA GEMM ----------------
// out[o, p] = sum_{i,k} W[o, i*9+k] * ( sum_n vals[k,p,n] * u[i, idx[p,n]] )
// One block = one 16-pixel tile; one wave per 16-row M tile (wave32).
__global__ void disco_wmma_kernel(
    const float* __restrict__ uin, const int* __restrict__ idx,
    const float* __restrict__ vals, const unsigned short* __restrict__ Wb,
    float* __restrict__ outF,
    int Cin, int Pin, int Pout, int Kvalid, int Kpad, int Mvalid)
{
  extern __shared__ unsigned char smem[];
  float* xg   = (float*)smem;                        // [Cin][6][16]
  float* valt = xg + Cin * 96;                       // [9][16][6]
  unsigned short* zt = (unsigned short*)(valt + 864);// fragment-order tile

  const int tid  = threadIdx.x;
  const int nthr = blockDim.x;
  const int p0   = blockIdx.x * 16;

  // stage gathered inputs
  for (int t = tid; t < Cin * 96; t += nthr) {
    int c = t / 96, r = t % 96, n = r / 16, pl = r % 16;
    int p = p0 + pl; if (p >= Pout) p = Pout - 1;
    xg[t] = uin[c * Pin + idx[p * 6 + n]];
  }
  // stage psi values
  for (int t = tid; t < 864; t += nthr) {
    int k = t / 96, r = t % 96, pl = r / 6, n = r % 6;
    int p = p0 + pl; if (p >= Pout) p = Pout - 1;
    valt[t] = vals[(k * Pout + p) * 6 + n];
  }
  __syncthreads();

  // build Z tile (bf16) in fragment order, zero-padded K rows
  for (int t = tid; t < Kpad * 16; t += nthr) {
    int row = t >> 4, pl = t & 15;
    float s = 0.0f;
    if (row < Kvalid) {
      int c = row / 9, k = row % 9;
      const float* vk = valt + (k * 16 + pl) * 6;
      const float* xc = xg + c * 96 + pl;
      #pragma unroll
      for (int n = 0; n < 6; ++n) s += vk[n] * xc[n * 16];
    }
    zt[frag_index(row, pl)] = f2bf(s);
  }
  __syncthreads();

  const int lane = tid & 31;
  const int wave = tid >> 5;
  const int m0   = wave * 16;
  const int mrow = lane & 15;
  const bool hi  = lane >= 16;
  const int hiA  = hi ? 8 : 0;

  const unsigned short* Arow = Wb + (m0 + mrow) * Kpad;
  __builtin_prefetch(Arow, 0, 1);   // global_prefetch_b8: pull weight row toward L2/L0

  v8f acc;
  #pragma unroll
  for (int r = 0; r < 8; ++r) acc[r] = 0.0f;

  for (int kk = 0; kk < Kpad; kk += 32) {
    BF16Frag a, b;
    const v4u* Ap = (const v4u*)(Arow + kk + hiA);   // 16B aligned: Kpad%32==0
    a.q[0] = Ap[0];            // K = kk+hiA .. +7
    a.q[1] = Ap[2];            // K = kk+16+hiA .. +23
    const v4u* Bp = (const v4u*)zt + (size_t)(((kk >> 5) * 32) + lane) * 2;
    b.q[0] = Bp[0];
    b.q[1] = Bp[1];
    acc = __builtin_amdgcn_wmma_f32_16x16x32_bf16(false, a.v, false, b.v,
                                                  (short)0, acc, false, false);
  }

  const int col = p0 + mrow;
  if (col < Pout) {
    #pragma unroll
    for (int r = 0; r < 8; ++r) {
      int row = m0 + r + (hi ? 8 : 0);
      if (row < Mvalid) outF[row * Pout + col] = acc[r];
    }
  }
}

// ---------------- generic WMMA GEMM with epilogues ----------------
// C(M x N) = A(M x Kpad, bf16) * B(Kpad x N, bf16). One block = one 16-col tile,
// all M tiles (one wave each); B tile staged to LDS in fragment order.
// mode 0: outF = acc ; mode 1: outB = bf16(gelu(acc)) ; mode 2: outF = res + ls[row]*acc
__global__ void gemm_wmma_kernel(
    const unsigned short* __restrict__ A, const unsigned short* __restrict__ B,
    const float* res, const float* ls,
    float* outF, unsigned short* outB,
    int Kpad, int N, int mode)
{
  extern __shared__ unsigned char smem[];
  unsigned short* bt = (unsigned short*)smem;        // fragment-order B tile

  const int tid  = threadIdx.x;
  const int nthr = blockDim.x;
  const int p0   = blockIdx.x * 16;

  for (int t = tid; t < Kpad * 16; t += nthr) {
    int row = t >> 4, pl = t & 15;
    int col = p0 + pl; if (col >= N) col = N - 1;    // clamp (stores masked later)
    bt[frag_index(row, pl)] = B[row * N + col];
  }
  __syncthreads();

  const int lane = tid & 31;
  const int wave = tid >> 5;
  const int m0   = wave * 16;
  const int mrow = lane & 15;
  const bool hi  = lane >= 16;
  const int hiA  = hi ? 8 : 0;

  const unsigned short* Arow = A + (m0 + mrow) * Kpad;
  __builtin_prefetch(Arow, 0, 1);

  v8f acc;
  #pragma unroll
  for (int r = 0; r < 8; ++r) acc[r] = 0.0f;

  for (int kk = 0; kk < Kpad; kk += 32) {
    BF16Frag a, b;
    const v4u* Ap = (const v4u*)(Arow + kk + hiA);
    a.q[0] = Ap[0];
    a.q[1] = Ap[2];
    const v4u* Bp = (const v4u*)bt + (size_t)(((kk >> 5) * 32) + lane) * 2;
    b.q[0] = Bp[0];
    b.q[1] = Bp[1];
    acc = __builtin_amdgcn_wmma_f32_16x16x32_bf16(false, a.v, false, b.v,
                                                  (short)0, acc, false, false);
  }

  const int nc = p0 + mrow;
  if (nc < N) {
    #pragma unroll
    for (int r = 0; r < 8; ++r) {
      int row = m0 + r + (hi ? 8 : 0);
      int o = row * N + nc;
      if (mode == 0)      outF[o] = acc[r];
      else if (mode == 1) outB[o] = f2bf(gelu_tanh(acc[r]));
      else                outF[o] = res[o] + ls[row] * acc[r];
    }
  }
}

// ---------------- packing / conversion ----------------
__global__ void pack_bf16_kernel(const float* __restrict__ src,
                                 unsigned short* __restrict__ dst,
                                 int Mv, int Kv, int Kp, int total)
{
  int t = blockIdx.x * blockDim.x + threadIdx.x;
  if (t >= total) return;
  int m = t / Kp, k = t % Kp;
  float v = (m < Mv && k < Kv) ? src[m * Kv + k] : 0.0f;
  dst[t] = f2bf(v);
}

__global__ void tobf16_rows_kernel(const float* __restrict__ src,
                                   unsigned short* __restrict__ dst,
                                   int rowsValid, int N, int total)
{
  int t = blockIdx.x * blockDim.x + threadIdx.x;
  if (t >= total) return;
  int r = t / N;
  dst[t] = (r < rowsValid) ? f2bf(src[t]) : (unsigned short)0;
}

// ---------------- spectral path (fp32, DFT as matmul with tables) ------------
__global__ void init_tables_kernel(float* Fc, float* Fs, float* iFc, float* iFs)
{
  int t = blockIdx.x * blockDim.x + threadIdx.x;
  if (t >= 90 * 180) return;
  int m = t / 180, w = t % 180;
  float ang = 6.283185307179586f * (float)(m * w) / 180.0f;
  float c = cosf(ang), s = sinf(ang);
  Fc[m * 180 + w] = c / 180.0f;
  Fs[m * 180 + w] = -s / 180.0f;
  float wgt = (m == 0) ? 1.0f : 2.0f;      // irfft(full*WP) folding
  iFc[w * 90 + m] = wgt * c;
  iFs[w * 90 + m] = -wgt * s;
}

__global__ void dft_fwd_kernel(const float* __restrict__ g,
                               const float* __restrict__ Fc, const float* __restrict__ Fs,
                               float* __restrict__ xre, float* __restrict__ xim)
{
  int t = blockIdx.x * blockDim.x + threadIdx.x;
  if (t >= 4320 * 90) return;
  int row = t / 90, m = t % 90;
  const float* gr = g + row * 180;
  const float* fc = Fc + m * 180;
  const float* fs = Fs + m * 180;
  float re = 0.f, im = 0.f;
  for (int w = 0; w < 180; ++w) { float v = gr[w]; re += v * fc[w]; im += v * fs[w]; }
  xre[row * 90 + m] = re; xim[row * 90 + m] = im;
}

__global__ void sht_kernel(const float* __restrict__ Pw,
                           const float* __restrict__ xre, const float* __restrict__ xim,
                           float* __restrict__ cre, float* __restrict__ cim)
{
  int t = blockIdx.x * blockDim.x + threadIdx.x;
  if (t >= CE * 8100) return;
  int c = t / 8100, r = t % 8100, l = r / 90, m = r % 90;
  const float* pw = Pw + (l * 90 + m) * 90;
  const float* xr = xre + c * 8100 + m;
  const float* xi = xim + c * 8100 + m;
  float re = 0.f, im = 0.f;
  for (int tt = 0; tt < 90; ++tt) { float p = pw[tt]; re += p * xr[tt * 90]; im += p * xi[tt * 90]; }
  cre[c * 8100 + l * 90 + m] = re;
  cim[c * 8100 + l * 90 + m] = im;
}

__global__ void gw_kernel(const float* __restrict__ gwp,
                          const float* __restrict__ cre, const float* __restrict__ cim,
                          float* __restrict__ ore, float* __restrict__ oim)
{
  int t = blockIdx.x * blockDim.x + threadIdx.x;
  if (t >= CE * 8100) return;
  int o = t / 8100, r = t % 8100, l = r / 90, m = r % 90;
  const float* wrow = gwp + o * 4320 + l;   // stride over i = 90
  float re = 0.f, im = 0.f;
  for (int i = 0; i < CE; ++i) {
    float wv = wrow[i * 90];
    re += wv * cre[i * 8100 + l * 90 + m];
    im += wv * cim[i * 8100 + l * 90 + m];
  }
  ore[o * 8100 + l * 90 + m] = re;
  oim[o * 8100 + l * 90 + m] = im;
}

__global__ void isht_kernel(const float* __restrict__ Pi,
                            const float* __restrict__ cre, const float* __restrict__ cim,
                            float* __restrict__ xre, float* __restrict__ xim)
{
  int t = blockIdx.x * blockDim.x + threadIdx.x;
  if (t >= CE * 8100) return;
  int c = t / 8100, r = t % 8100, tt = r / 90, m = r % 90;
  const float* cr = cre + c * 8100 + m;
  const float* ci = cim + c * 8100 + m;
  const float* pi = Pi + m * 90 + tt;       // stride over l = 8100
  float re = 0.f, im = 0.f;
  for (int l = 0; l < 90; ++l) { float p = pi[l * 8100]; re += p * cr[l * 90]; im += p * ci[l * 90]; }
  xre[c * 8100 + tt * 90 + m] = re;
  xim[c * 8100 + tt * 90 + m] = im;
}

__global__ void dft_inv_kernel(const float* __restrict__ xre, const float* __restrict__ xim,
                               const float* __restrict__ iFc, const float* __restrict__ iFs,
                               float* __restrict__ gout)
{
  int t = blockIdx.x * blockDim.x + threadIdx.x;
  if (t >= 4320 * 180) return;
  int row = t / 180, w = t % 180;
  const float* xr = xre + row * 90;
  const float* xi = xim + row * 90;
  const float* fc = iFc + w * 90;
  const float* fs = iFs + w * 90;
  float s = 0.f;
  for (int m = 0; m < 90; ++m) s += xr[m] * fc[m] + xi[m] * fs[m];
  gout[row * 180 + w] = s;
}

// ---------------- bilinear upsample 90x180 -> 361x720 ----------------
__global__ void bilinear_kernel(const float* __restrict__ g,
    const int* __restrict__ li0, const int* __restrict__ li1, const float* __restrict__ lwv,
    const int* __restrict__ ji0, const int* __restrict__ ji1, const float* __restrict__ jwv,
    float* __restrict__ uout)
{
  int t = blockIdx.x * blockDim.x + threadIdx.x;
  if (t >= CE * HFWF) return;
  int c = t / HFWF, r = t % HFWF, i = r / 720, j = r % 720;
  const float* gc = g + c * HPWP;
  int a0 = li0[i], a1 = li1[i]; float aw = lwv[i];
  int b0 = ji0[j], b1 = ji1[j]; float bw = jwv[j];
  float t0 = gc[a0 * 180 + b0] * (1.f - aw) + gc[a1 * 180 + b0] * aw;
  float t1 = gc[a0 * 180 + b1] * (1.f - aw) + gc[a1 * 180 + b1] * aw;
  uout[t] = t0 * (1.f - bw) + t1 * bw;
}

// ---------------- host orchestration ----------------
extern "C" void kernel_launch(void* const* d_in, const int* in_sizes, int n_in,
                              void* d_out, int out_size, void* d_ws, size_t ws_size,
                              hipStream_t stream) {
  (void)in_sizes; (void)n_in; (void)out_size; (void)ws_size;

  const float* x        = (const float*)d_in[0];
  const int*   enc_idx  = (const int*)  d_in[1];
  const float* enc_vals = (const float*)d_in[2];
  const float* enc_w    = (const float*)d_in[3];
  const int*   proc_idx = (const int*)  d_in[4];
  const float* proc_vals= (const float*)d_in[5];
  const float* lw       = (const float*)d_in[6];
  const float* gw       = (const float*)d_in[7];
  const float* Pw       = (const float*)d_in[8];
  const float* Pi       = (const float*)d_in[9];
  const float* m1       = (const float*)d_in[10];
  const float* m2       = (const float*)d_in[11];
  const float* ls       = (const float*)d_in[12];
  const int*   lat_i0   = (const int*)  d_in[13];
  const int*   lat_i1   = (const int*)  d_in[14];
  const float* lat_w    = (const float*)d_in[15];
  const int*   lon_i0   = (const int*)  d_in[16];
  const int*   lon_i1   = (const int*)  d_in[17];
  const float* lon_w    = (const float*)d_in[18];
  const int*   dec_idx  = (const int*)  d_in[19];
  const float* dec_vals = (const float*)d_in[20];
  const float* dec_w    = (const float*)d_in[21];

  unsigned char* ws = (unsigned char*)d_ws;
  size_t off = 0;
  auto alloc = [&](size_t bytes) -> void* {
    off = (off + 255) & ~(size_t)255;
    void* p = ws + off; off += bytes; return p;
  };

  float* h      = (float*)alloc((size_t)CE * HPWP * 4);
  float* dx     = (float*)alloc((size_t)CE * HPWP * 4);
  unsigned short* dxbf  = (unsigned short*)alloc((size_t)64 * HPWP * 2);
  unsigned short* actbf = (unsigned short*)alloc((size_t)HD * HPWP * 2);
  float* ubuf   = (float*)alloc((size_t)CE * HFWF * 4);
  float* spA_re = (float*)alloc((size_t)CE * 8100 * 4);
  float* spA_im = (float*)alloc((size_t)CE * 8100 * 4);
  float* spB_re = (float*)alloc((size_t)CE * 8100 * 4);
  float* spB_im = (float*)alloc((size_t)CE * 8100 * 4);
  float* Fc  = (float*)alloc(16200 * 4);
  float* Fs  = (float*)alloc(16200 * 4);
  float* iFc = (float*)alloc(16200 * 4);
  float* iFs = (float*)alloc(16200 * 4);
  unsigned short* enc_wb = (unsigned short*)alloc((size_t)48 * 256 * 2);
  unsigned short* lw_b   = (unsigned short*)alloc((size_t)2 * 48 * 448 * 2);
  unsigned short* dec_wb = (unsigned short*)alloc((size_t)32 * 448 * 2);
  unsigned short* m1_b   = (unsigned short*)alloc((size_t)4 * 96 * 64 * 2);
  unsigned short* m2_b   = (unsigned short*)alloc((size_t)4 * 48 * 96 * 2);

  const int B256 = 256;
  #define GRID1(n) dim3(((n) + B256 - 1) / B256)

  // DFT tables
  init_tables_kernel<<<GRID1(16200), B256, 0, stream>>>(Fc, Fs, iFc, iFs);

  // weight packing (fp32 -> zero-padded bf16)
  pack_bf16_kernel<<<GRID1(48 * 256), B256, 0, stream>>>(enc_w, enc_wb, 48, 234, 256, 48 * 256);
  for (int hh = 0; hh < 2; ++hh)
    pack_bf16_kernel<<<GRID1(48 * 448), B256, 0, stream>>>(lw + hh * 20736, lw_b + hh * 48 * 448,
                                                           48, 432, 448, 48 * 448);
  pack_bf16_kernel<<<GRID1(32 * 448), B256, 0, stream>>>(dec_w, dec_wb, 24, 432, 448, 32 * 448);
  for (int i = 0; i < 4; ++i) {
    pack_bf16_kernel<<<GRID1(96 * 64), B256, 0, stream>>>(m1 + i * 96 * 48, m1_b + i * 96 * 64,
                                                          96, 48, 64, 96 * 64);
    pack_bf16_kernel<<<GRID1(48 * 96), B256, 0, stream>>>(m2 + i * 48 * 96, m2_b + i * 48 * 96,
                                                          48, 96, 96, 48 * 96);
  }

  const size_t ldsEnc = (size_t)26 * 96 * 4 + 864 * 4 + 256 * 16 * 2;
  const size_t ldsPrc = (size_t)48 * 96 * 4 + 864 * 4 + 448 * 16 * 2;

  // encoder: x(26, 259920) -> h(48, 16200)
  disco_wmma_kernel<<<dim3((HPWP + 15) / 16), dim3(96), ldsEnc, stream>>>(
      x, enc_idx, enc_vals, enc_wb, h, 26, HFWF, HPWP, 234, 256, 48);

  for (int i = 0; i < 4; ++i) {
    if ((i & 1) == 0) {
      int hh = i / 2;
      dft_fwd_kernel<<<GRID1(4320 * 90), B256, 0, stream>>>(h, Fc, Fs, spA_re, spA_im);
      sht_kernel<<<GRID1(CE * 8100), B256, 0, stream>>>(Pw, spA_re, spA_im, spB_re, spB_im);
      gw_kernel<<<GRID1(CE * 8100), B256, 0, stream>>>(gw + hh * 194400, spB_re, spB_im, spA_re, spA_im);
      isht_kernel<<<GRID1(CE * 8100), B256, 0, stream>>>(Pi, spA_re, spA_im, spB_re, spB_im);
      dft_inv_kernel<<<GRID1(4320 * 180), B256, 0, stream>>>(spB_re, spB_im, iFc, iFs, dx);
    } else {
      int hh = i / 2;
      disco_wmma_kernel<<<dim3((HPWP + 15) / 16), dim3(96), ldsPrc, stream>>>(
          h, proc_idx, proc_vals, lw_b + hh * 48 * 448, dx, 48, HPWP, HPWP, 432, 448, 48);
    }
    // MLP: dx -> gelu(m1 dx) -> m2 -> residual into h
    tobf16_rows_kernel<<<GRID1(64 * HPWP), B256, 0, stream>>>(dx, dxbf, 48, HPWP, 64 * HPWP);
    gemm_wmma_kernel<<<dim3((HPWP + 15) / 16), dim3(32 * (HD / 16)), (size_t)64 * 16 * 2, stream>>>(
        m1_b + i * 96 * 64, dxbf, nullptr, nullptr, nullptr, actbf, 64, HPWP, 1);
    gemm_wmma_kernel<<<dim3((HPWP + 15) / 16), dim3(32 * (CE / 16)), (size_t)96 * 16 * 2, stream>>>(
        m2_b + i * 48 * 96, actbf, h, ls + i * CE, h, nullptr, 96, HPWP, 2);
  }

  // bilinear upsample h -> ubuf (48, 361*720)
  bilinear_kernel<<<GRID1(CE * HFWF), B256, 0, stream>>>(
      h, lat_i0, lat_i1, lat_w, lon_i0, lon_i1, lon_w, ubuf);

  // decoder: ubuf(48, 259920) -> out(24, 259920)
  disco_wmma_kernel<<<dim3(HFWF / 16), dim3(64), ldsPrc, stream>>>(
      ubuf, dec_idx, dec_vals, dec_wb, (float*)d_out, 48, HFWF, HFWF, 432, 448, COUT);
}
